// MemoryReader_16930761081020
// MI455X (gfx1250) — compile-verified, matrix operand order checked by reading, not run on hardware
//
#include <hip/hip_runtime.h>

#define B_    4
#define CK_   64
#define CV_   512
#define H_    30
#define W_    54
#define T_    8
#define P_    (H_*W_)          // 1620
#define N_    (T_*H_*W_)       // 12960
#define CHUNK 128
#define NCHUNK ((N_ + CHUNK - 1)/CHUNK)   // 102 (last chunk has 32 valid rows)
#define PTILES ((P_ + 15)/16)             // 102

typedef __attribute__((ext_vector_type(16))) __bf16       v16bf;
typedef __attribute__((ext_vector_type(8)))  float        v8f;
typedef __attribute__((ext_vector_type(4)))  unsigned int v4u;

union Frag16 {
  v16bf  bf;
  v4u    q[2];
  __bf16 h[16];
};
union Pack8 {
  v4u    q;
  __bf16 h[8];
};
union Pack4 {
  uint2  u;
  __bf16 h[4];
};

// LDS row strides (in bf16 elements); chosen so every b128 access is 16B
// aligned and row starts hit distinct bank groups.
#define QSTR  72
#define MKSTR 72
#define PSTR  136

template <bool MV_BF16>
__global__ __launch_bounds__(256) void mem_reader_attn(
    const float* __restrict__ mk, const float* __restrict__ mv_f32,
    const __bf16* __restrict__ mv_bf, const float* __restrict__ qk,
    float* __restrict__ out)
{
  __shared__ __bf16 s_q[16 * QSTR];        // q tile, [p][c], pre-scaled by 1/8
  __shared__ __bf16 s_mk[CHUNK * MKSTR];   // mk chunk, [n_local][c]
  __shared__ __bf16 s_prob[16 * PSTR];     // probabilities, [p][n_local]
  __shared__ float  s_pmax[8][32];
  __shared__ float  s_psum[8][32];
  __shared__ float  s_mrun[16];
  __shared__ float  s_lrun[16];
  __shared__ float  s_alpha[16];

  const int tid  = threadIdx.x;
  const int wave = tid >> 5;
  const int lane = tid & 31;
  const int col  = lane & 15;        // B-matrix column / C-matrix column / A row
  const int kg   = (lane >> 4) * 8;  // K-half base (A/B) and M-half base (C/D)
  const int p0   = blockIdx.x * 16;
  const int bi   = blockIdx.y;

  const float*  mk_b  = mk + (size_t)bi * CK_ * N_;
  const float*  qk_b  = qk + (size_t)bi * CK_ * P_;
  const float*  mvf_b = mv_f32 ? (mv_f32 + (size_t)bi * CV_ * N_) : nullptr;
  const __bf16* mvb_b = mv_bf  ? (mv_bf  + (size_t)bi * CV_ * N_) : nullptr;

  // ---- load q tile once: s_q[p][c] = bf16(qk[c, p0+p] / sqrt(CK)) ----
  for (int idx = tid; idx < CK_ * 16; idx += 256) {
    int p = idx & 15, c = idx >> 4;
    int pg = p0 + p; if (pg > P_ - 1) pg = P_ - 1;
    s_q[p * QSTR + c] = (__bf16)(qk_b[(size_t)c * P_ + pg] * 0.125f);
  }
  if (tid < 16) { s_mrun[tid] = -1e30f; s_lrun[tid] = 0.f; s_alpha[tid] = 0.f; }

  v8f acc[4] = {v8f{}, v8f{}, v8f{}, v8f{}};   // 64 CV rows per wave

  for (int nc = 0; nc < NCHUNK; ++nc) {
    const int n0 = nc * CHUNK;
    __syncthreads();   // previous iteration fully done with LDS

    // ---- stage mk chunk into LDS, transposed to [n][c], bf16, zero OOB ----
    for (int idx = tid; idx < CK_ * CHUNK; idx += 256) {
      int nl = idx & (CHUNK - 1), c = idx >> 7;
      int ng = n0 + nl;
      float v = (ng < N_) ? mk_b[(size_t)c * N_ + ng] : 0.f;
      s_mk[nl * MKSTR + c] = (__bf16)v;
    }
    __syncthreads();

    // ---- stage A: S[16n x 16p] per wave, K = CK = 64 (2 k-steps) ----
    const int mrow = wave * 16;
    v8f S = {};
#pragma unroll
    for (int ks = 0; ks < 2; ++ks) {
      Frag16 a, b;
      const __bf16* ap = &s_mk[(mrow + col) * MKSTR + ks * 32 + kg];
      a.q[0] = *(const v4u*)ap;  a.q[1] = *(const v4u*)(ap + 16);
      const __bf16* bp = &s_q[col * QSTR + ks * 32 + kg];
      b.q[0] = *(const v4u*)bp;  b.q[1] = *(const v4u*)(bp + 16);
      S = __builtin_amdgcn_wmma_f32_16x16x32_bf16(false, a.bf, false, b.bf,
                                                  (short)0, S, false, false);
    }

    // per-lane column max over this wave's 8 rows (mask OOB n rows)
    float sv[8];
    float lmax = -1e30f;
#pragma unroll
    for (int j = 0; j < 8; ++j) {
      int ng = n0 + mrow + kg + j;
      sv[j] = (ng < N_) ? S[j] : -1e30f;
      lmax = fmaxf(lmax, sv[j]);
    }
    s_pmax[wave][lane] = lmax;
    __syncthreads();

    if (tid < 16) {   // running max + rescale factor per query column
      float cmax = -1e30f;
#pragma unroll
      for (int w = 0; w < 8; ++w)
        cmax = fmaxf(cmax, fmaxf(s_pmax[w][tid], s_pmax[w][tid + 16]));
      float mold = s_mrun[tid];
      float mnew = fmaxf(mold, cmax);
      s_alpha[tid] = __expf(mold - mnew);
      s_mrun[tid]  = mnew;
    }
    __syncthreads();

    const float mnew = s_mrun[col];
    const float alph = s_alpha[col];
#pragma unroll
    for (int t = 0; t < 4; ++t) acc[t] *= alph;   // online rescale

    // probabilities -> bf16, packed b128 store into s_prob[p][n]
    Pack8 pk;
    float lsum = 0.f;
#pragma unroll
    for (int j = 0; j < 8; ++j) {
      int ng = n0 + mrow + kg + j;
      float pr = (ng < N_) ? __expf(sv[j] - mnew) : 0.f;
      lsum += pr;
      pk.h[j] = (__bf16)pr;
    }
    *(v4u*)&s_prob[col * PSTR + mrow + kg] = pk.q;
    s_psum[wave][lane] = lsum;
    __syncthreads();

    if (tid < 16) {   // running sum per query column
      float csum = 0.f;
#pragma unroll
      for (int w = 0; w < 8; ++w)
        csum += s_psum[w][tid] + s_psum[w][tid + 16];
      s_lrun[tid] = s_lrun[tid] * s_alpha[tid] + csum;
    }

    // ---- stage B: acc[v,p] += mv[v, chunk] x prob[chunk, p] ----
#pragma unroll
    for (int vt = 0; vt < 4; ++vt) {
      const int v = wave * 64 + vt * 16 + col;     // A-matrix row (this lane)
#pragma unroll
      for (int ks = 0; ks < 4; ++ks) {
        int nA = n0 + ks * 32 + kg;
        int b1 = nA;      if (b1 > N_ - 8) b1 = N_ - 8;   // clamp (prob==0 there)
        int b2 = nA + 16; if (b2 > N_ - 8) b2 = N_ - 8;
        Frag16 a;
        if (MV_BF16) {
          // pre-converted bf16 mv: one b128 per K-half, no cvt on the WMMA path
          const __bf16* rowp = mvb_b + (size_t)v * N_;
          a.q[0] = *(const v4u*)(rowp + b1);
          a.q[1] = *(const v4u*)(rowp + b2);
        } else {
          const float* rowp = mvf_b + (size_t)v * N_;
          const float4 f0 = *(const float4*)(rowp + b1);
          const float4 f1 = *(const float4*)(rowp + b1 + 4);
          const float4 f2 = *(const float4*)(rowp + b2);
          const float4 f3 = *(const float4*)(rowp + b2 + 4);
          a.h[0]=(__bf16)f0.x;  a.h[1]=(__bf16)f0.y;  a.h[2]=(__bf16)f0.z;  a.h[3]=(__bf16)f0.w;
          a.h[4]=(__bf16)f1.x;  a.h[5]=(__bf16)f1.y;  a.h[6]=(__bf16)f1.z;  a.h[7]=(__bf16)f1.w;
          a.h[8]=(__bf16)f2.x;  a.h[9]=(__bf16)f2.y;  a.h[10]=(__bf16)f2.z; a.h[11]=(__bf16)f2.w;
          a.h[12]=(__bf16)f3.x; a.h[13]=(__bf16)f3.y; a.h[14]=(__bf16)f3.z; a.h[15]=(__bf16)f3.w;
        }
        Frag16 b;
        const __bf16* bp = &s_prob[col * PSTR + ks * 32 + kg];
        b.q[0] = *(const v4u*)bp;  b.q[1] = *(const v4u*)(bp + 16);
        acc[vt] = __builtin_amdgcn_wmma_f32_16x16x32_bf16(false, a.bf, false, b.bf,
                                                          (short)0, acc[vt], false, false);
      }
    }
  }

  // ---- epilogue: divide by softmax denominator, store mem part ----
  __syncthreads();
  const float linv = 1.0f / s_lrun[col];
  const int pg = p0 + col;
  if (pg < P_) {
#pragma unroll
    for (int vt = 0; vt < 4; ++vt) {
#pragma unroll
      for (int j = 0; j < 8; ++j) {
        int v = wave * 64 + vt * 16 + kg + j;
        out[((size_t)bi * (2 * CV_) + v) * P_ + pg] = acc[vt][j] * linv;
      }
    }
  }
}

// Streaming f32 -> bf16 conversion of mv into workspace (one pass, warms L2).
__global__ __launch_bounds__(256) void convert_mv(const float* __restrict__ src,
                                                  __bf16* __restrict__ dst, int n4)
{
  int i = blockIdx.x * blockDim.x + threadIdx.x;
  if (i < n4) {
    float4 f = ((const float4*)src)[i];
    Pack4 p;
    p.h[0] = (__bf16)f.x; p.h[1] = (__bf16)f.y;
    p.h[2] = (__bf16)f.z; p.h[3] = (__bf16)f.w;
    *(uint2*)(dst + 4 * (size_t)i) = p.u;
  }
}

// out[b, CV + c, p] = qv[b, c, p]
__global__ __launch_bounds__(256) void qv_copy(const float* __restrict__ qv,
                                               float* __restrict__ out, int total)
{
  int i = blockIdx.x * blockDim.x + threadIdx.x;
  if (i < total) {
    int b = i / (CV_ * P_);
    int r = i - b * (CV_ * P_);
    out[(size_t)b * (2 * CV_ * P_) + (CV_ * P_) + r] = qv[i];
  }
}

extern "C" void kernel_launch(void* const* d_in, const int* in_sizes, int n_in,
                              void* d_out, int out_size, void* d_ws, size_t ws_size,
                              hipStream_t stream) {
  const float* mk = (const float*)d_in[0];
  const float* mv = (const float*)d_in[1];
  const float* qk = (const float*)d_in[2];
  const float* qv = (const float*)d_in[3];
  float* out = (float*)d_out;

  const size_t mv_elems = (size_t)B_ * CV_ * N_;
  const size_t need     = mv_elems * sizeof(__bf16);
  dim3 grid(PTILES, B_);

  if (d_ws && ws_size >= need) {
    __bf16* mv_bf = (__bf16*)d_ws;
    int n4 = (int)(mv_elems / 4);
    convert_mv<<<(n4 + 255) / 256, 256, 0, stream>>>(mv, mv_bf, n4);
    mem_reader_attn<true><<<grid, 256, 0, stream>>>(mk, nullptr, mv_bf, qk, out);
  } else {
    mem_reader_attn<false><<<grid, 256, 0, stream>>>(mk, mv, nullptr, qk, out);
  }

  int total = B_ * CV_ * P_;
  qv_copy<<<(total + 255) / 256, 256, 0, stream>>>(qv, out, total);
}
